// Layer_88167088652319
// MI455X (gfx1250) — compile-verified, matrix operand order checked by reading, not run on hardware
//
#include <hip/hip_runtime.h>

// Problem constants (from reference setup_inputs)
#define DD 512
#define G3 1536          // 3*D
#define BB 2
#define TT 256
#define MM 80
#define NROWS (BB*TT*MM) // 40960 token rows

typedef __attribute__((ext_vector_type(16))) __bf16          v16bf;
typedef __attribute__((ext_vector_type(16))) unsigned short  v16u;
typedef __attribute__((ext_vector_type(8)))  float           v8f;

// ---------- helpers ----------
// Hardware f32 -> bf16 convert (RNE) via clang cast; returns raw bits.
__device__ __forceinline__ unsigned short f2bf(float f) {
  __bf16 h = (__bf16)f;
  return __builtin_bit_cast(unsigned short, h);
}

// Fragment from bf16 (raw u16) source (global or LDS), 32-elem K panel.
// Per ISA 7.12.2: lanes 0-15 -> row=lane,    K = {k0..k0+7, k0+16..k0+23}
//                 lanes16-31 -> row=lane-16, K = {k0+8..k0+15, k0+24..k0+31}
// caller passes p already offset by (row*stride + koff), koff = (lane>>4)*8.
__device__ __forceinline__ v16bf frag_bf16(const unsigned short* p) {
  v16u t;
#pragma unroll
  for (int i = 0; i < 8; ++i) t[i] = p[i];
#pragma unroll
  for (int i = 0; i < 8; ++i) t[8 + i] = p[16 + i];
  return __builtin_bit_cast(v16bf, t);
}

// ---------- elementwise kernels ----------
__global__ void cvt_bf16_kernel(const float* __restrict__ s, unsigned short* __restrict__ d, int n) {
  int i = blockIdx.x * blockDim.x + threadIdx.x;
  if (i < n) d[i] = f2bf(s[i]);
}

__global__ void zero_kernel(float* __restrict__ p, int n) {
  int i = blockIdx.x * blockDim.x + threadIdx.x;
  if (i < n) p[i] = 0.0f;
}

__global__ void add2_kernel(const float* __restrict__ a, const float* __restrict__ b,
                            float* __restrict__ o, int n) {
  int i = blockIdx.x * blockDim.x + threadIdx.x;
  if (i < n) o[i] = a[i] + b[i];
}

// time_input[bm][t][d] = x_time[b][t][m][d]
__global__ void permute_time_kernel(const float* __restrict__ x, float* __restrict__ o, int n) {
  int i = blockIdx.x * blockDim.x + threadIdx.x;
  if (i >= n) return;
  int d   = i & (DD - 1);
  int row = i >> 9;          // bm*TT + t
  int bm  = row >> 8;        // /TT (TT==256)
  int t   = row & (TT - 1);
  int b   = bm / MM, m = bm % MM;
  size_t src = ((size_t)((b * TT + t) * MM + m) << 9) + d;
  o[i] = x[src];
}

// cat[r][0:512]=x1 (time GRU out, gathered), cat[r][512:1024]=y2, cat[r][1024:1536]=y3
__global__ void build_cat_kernel(const float* __restrict__ y1, const float* __restrict__ y2,
                                 const float* __restrict__ y3, float* __restrict__ cat, int n) {
  int i = blockIdx.x * blockDim.x + threadIdx.x;
  if (i >= n) return;
  int d = i & (DD - 1);
  int r = i >> 9;                       // r = (b*TT + t)*MM + m
  int b = r / (TT * MM);
  int rem = r - b * (TT * MM);
  int t = rem / MM, m = rem - t * MM;
  int bm = b * MM + m;
  size_t base = (size_t)r * G3;
  cat[base + d]        = y1[((size_t)(bm * TT + t) << 9) + d];
  cat[base + 512 + d]  = y2[((size_t)r << 9) + d];
  cat[base + 1024 + d] = y3[((size_t)r << 9) + d];
}

// ---------- blocked WMMA GEMM ----------
// C[r][n] = sum_k A[r][k]*B[n][k] + bias[n] (+ resid[r][n])
// A fp32 row-major (lda), B bf16 row-major (N x K).
// Block tile 128M x 128N, 256 threads = 8 waves; wave w computes rows
// [m0+16w, m0+16w+16) x all 128 N columns (8 accumulators).
// Per 32-K step: A (cvt f32->bf16) staged via VALU, B staged via
// GLOBAL_LOAD_ASYNC_TO_LDS_B128 (ASYNCcnt) so the B DMA overlaps the A convert.
template <bool HAS_RESID>
__global__ void gemm_bf16_kernel(const float* __restrict__ A, int lda,
                                 const unsigned short* __restrict__ B,
                                 const float* __restrict__ bias,
                                 const float* __restrict__ resid,
                                 float* __restrict__ C, int ldc, int K) {
  __shared__ unsigned short As[128 * 32];
  __shared__ unsigned short Bs[128 * 32];
  int tid  = threadIdx.x;
  int wave = tid >> 5, lane = tid & 31;
  int m0 = blockIdx.x << 7, n0 = blockIdx.y << 7;
  int koff = (lane >> 4) << 3;       // 0 or 8 (K-half select)
  int rl   = lane & 15;              // row-in-tile

  v8f zero = {};
  v8f acc[8];
#pragma unroll
  for (int j = 0; j < 8; ++j) acc[j] = zero;

  for (int k0 = 0; k0 < K; k0 += 32) {
    // --- B slab: async DMA global -> LDS, 512 x b128, 2 per thread ---
#pragma unroll
    for (int i = 0; i < 2; ++i) {
      int q   = tid + (i << 8);                    // 0..511 b128 chunks
      int row = q >> 2;                            // 4 chunks per 32-elem row
      int c8  = (q & 3) << 3;
      const unsigned short* gsrc = B + (size_t)(n0 + row) * K + (k0 + c8);
      unsigned ldst = (unsigned)(unsigned long long)(&Bs[q << 3]);
      asm volatile("global_load_async_to_lds_b128 %0, %1, off"
                   :: "v"(ldst), "v"(gsrc) : "memory");
    }
    // --- A slab: load f32, convert to bf16, store to LDS (overlaps B DMA) ---
#pragma unroll
    for (int i = 0; i < 16; ++i) {
      int e = tid + (i << 8);        // 0..4095
      int r = e >> 5, c = e & 31;
      As[e] = f2bf(A[(size_t)(m0 + r) * lda + (k0 + c)]);
    }
    asm volatile("s_wait_asynccnt 0x0" ::: "memory");
    __syncthreads();

    v16bf a = frag_bf16(&As[(wave * 16 + rl) * 32 + koff]);
#pragma unroll
    for (int j = 0; j < 8; ++j) {
      v16bf b = frag_bf16(&Bs[(j * 16 + rl) * 32 + koff]);
      acc[j] = __builtin_amdgcn_wmma_f32_16x16x32_bf16(false, a, false, b, (short)0, acc[j],
                                                       false, false);
    }
    __syncthreads();
  }

  // epilogue: C layout VGPR r -> M = r (lanes 0-15) / r+8 (lanes 16-31); N = lane%16
  int mb = m0 + wave * 16 + koff;
#pragma unroll
  for (int j = 0; j < 8; ++j) {
    int n = n0 + j * 16 + rl;
    float bv = bias[n];
#pragma unroll
    for (int r = 0; r < 8; ++r) {
      size_t off = (size_t)(mb + r) * ldc + n;
      float v = acc[j][r] + bv;
      if (HAS_RESID) v += resid[off];
      C[off] = v;
    }
  }
}

// ---------- fused GRU step: gh = h @ Whh^T + bhh (WMMA via LDS), then gates ----------
// grid.x = NB/16 (16-row slab per block), block = 512 threads (16 waves).
// dynamic LDS: 16*512 bf16 h-slab + 16*1536 f32 gh = 114688 bytes.
__global__ void gru_step_kernel(float* __restrict__ h,                    // (NB,512) in/out
                                const unsigned short* __restrict__ Whh,  // (1536,512) bf16
                                const float* __restrict__ bhh,           // (1536)
                                const float* __restrict__ gi,            // (NB, L, 1536)
                                float* __restrict__ y,                   // (NB, L, 512)
                                int L, int step) {
  extern __shared__ char smem[];
  unsigned short* hs = (unsigned short*)smem;              // 16 x 512 bf16
  float* ghs = (float*)(smem + 16 * DD * 2);               // 16 x 1536 f32
  int row0 = blockIdx.x << 4;
  int tid = threadIdx.x;

  // stage h slab -> LDS (bf16)
  for (int e = tid; e < 16 * DD; e += 512) {
    int r = e >> 9, c = e & (DD - 1);
    hs[e] = f2bf(h[((size_t)(row0 + r) << 9) + c]);
  }
  __syncthreads();

  int wave = tid >> 5, lane = tid & 31;
  int koff = (lane >> 4) << 3;
  int rl   = lane & 15;
  // 96 column tiles of 16 over 1536 outputs, 6 per wave
#pragma unroll
  for (int it = 0; it < 6; ++it) {
    int n0 = (wave * 6 + it) << 4;
    v8f acc = {};
    for (int k0 = 0; k0 < DD; k0 += 32) {
      v16bf a = frag_bf16(hs + (rl << 9) + k0 + koff);                    // ds reads
      v16bf b = frag_bf16(Whh + (size_t)(n0 + rl) * DD + k0 + koff);      // L2-hot weights
      acc = __builtin_amdgcn_wmma_f32_16x16x32_bf16(false, a, false, b, (short)0, acc,
                                                    false, false);
    }
    int n  = n0 + rl;
    int mb = koff;
    float bv = bhh[n];
#pragma unroll
    for (int r = 0; r < 8; ++r) ghs[(mb + r) * G3 + n] = acc[r] + bv;
  }
  __syncthreads();

  // gate math over 16 rows x 512 cols
  for (int e = tid; e < 16 * DD; e += 512) {
    int r = e >> 9, c = e & (DD - 1);
    size_t grow = (size_t)(row0 + r) * L + step;
    const float* gip = gi + grow * G3;
    float hr = ghs[r * G3 + c];
    float hz = ghs[r * G3 + 512 + c];
    float hn = ghs[r * G3 + 1024 + c];
    float rg = 1.0f / (1.0f + __expf(-(gip[c] + hr)));
    float zg = 1.0f / (1.0f + __expf(-(gip[512 + c] + hz)));
    float ng = tanhf(gip[1024 + c] + rg * hn);
    size_t hoff = ((size_t)(row0 + r) << 9) + c;
    float hnew = (1.0f - zg) * ng + zg * h[hoff];
    h[hoff] = hnew;
    y[(grow << 9) + c] = hnew;
  }
}

// ---------- host orchestration ----------
extern "C" void kernel_launch(void* const* d_in, const int* in_sizes, int n_in,
                              void* d_out, int out_size, void* d_ws, size_t ws_size,
                              hipStream_t stream) {
  (void)in_sizes; (void)n_in; (void)out_size; (void)ws_size;
  const float* x_time = (const float*)d_in[0];
  const float* x_freq = (const float*)d_in[1];
  const float* w_ih[4] = {(const float*)d_in[2], (const float*)d_in[6],  (const float*)d_in[10], (const float*)d_in[14]};
  const float* w_hh[4] = {(const float*)d_in[3], (const float*)d_in[7],  (const float*)d_in[11], (const float*)d_in[15]};
  const float* b_ih[4] = {(const float*)d_in[4], (const float*)d_in[8],  (const float*)d_in[12], (const float*)d_in[16]};
  const float* b_hh[4] = {(const float*)d_in[5], (const float*)d_in[9],  (const float*)d_in[13], (const float*)d_in[17]};
  const float* W_time = (const float*)d_in[18];
  const float* b_time = (const float*)d_in[19];
  const float* W_freq = (const float*)d_in[20];
  const float* b_freq = (const float*)d_in[21];

  // workspace carve-out
  char* ws = (char*)d_ws; size_t off = 0;
  auto alloc = [&](size_t bytes) -> char* {
    char* p = ws + off; off += (bytes + 255) & ~(size_t)255; return p;
  };
  unsigned short* wb_ih[4]; unsigned short* wb_hh[4];
  for (int i = 0; i < 4; ++i) {
    wb_ih[i] = (unsigned short*)alloc((size_t)G3 * DD * 2);
    wb_hh[i] = (unsigned short*)alloc((size_t)G3 * DD * 2);
  }
  unsigned short* wbWt = (unsigned short*)alloc((size_t)DD * G3 * 2);
  unsigned short* wbWf = (unsigned short*)alloc((size_t)DD * DD * 2);
  float* tinp = (float*)alloc((size_t)NROWS * DD * 4);
  float* gi   = (float*)alloc((size_t)NROWS * G3 * 4);
  float* y1   = (float*)alloc((size_t)NROWS * DD * 4);
  float* y2   = (float*)alloc((size_t)NROWS * DD * 4);
  float* y3   = (float*)alloc((size_t)NROWS * DD * 4);
  float* hb   = (float*)alloc((size_t)512 * DD * 4);
  float* xs   = tinp;   // tinp dead after gi_time GEMM
  float* ys   = y2;     // y2 dead after cat + time-out GEMM
  float* xt   = (float*)d_out;
  float* xf   = (float*)d_out + (size_t)NROWS * DD;

  const int EW = NROWS * DD;                  // 20,971,520 elementwise work items
  auto cvt = [&](const float* s, unsigned short* d, int n) {
    cvt_bf16_kernel<<<(n + 255) / 256, 256, 0, stream>>>(s, d, n);
  };
  for (int i = 0; i < 4; ++i) { cvt(w_ih[i], wb_ih[i], G3 * DD); cvt(w_hh[i], wb_hh[i], G3 * DD); }
  cvt(W_time, wbWt, DD * G3);
  cvt(W_freq, wbWf, DD * DD);

  permute_time_kernel<<<(EW + 255) / 256, 256, 0, stream>>>(x_time, tinp, EW);

  auto gemm = [&](const float* A, int lda, const unsigned short* Bm,
                  const float* bias, const float* resid,
                  float* C, int ldc, int R, int N, int K) {
    dim3 g(R >> 7, N >> 7);
    if (resid)
      gemm_bf16_kernel<true><<<g, 256, 0, stream>>>(A, lda, Bm, bias, resid, C, ldc, K);
    else
      gemm_bf16_kernel<false><<<g, 256, 0, stream>>>(A, lda, Bm, bias, nullptr, C, ldc, K);
  };

  const size_t shmem = (size_t)16 * DD * 2 + (size_t)16 * G3 * 4;  // 114688 B
  auto run_gru = [&](int g, float* yout, int NB, int L, bool rev) {
    zero_kernel<<<(NB * DD + 255) / 256, 256, 0, stream>>>(hb, NB * DD);
    for (int s = 0; s < L; ++s) {
      int step = rev ? (L - 1 - s) : s;
      gru_step_kernel<<<NB / 16, 512, shmem, stream>>>(hb, wb_hh[g], b_hh[g], gi, yout, L, step);
    }
  };

  // ---- TimeDelayedStack ----
  gemm(tinp,   DD, wb_ih[0], b_ih[0], nullptr, gi, G3, NROWS, G3, DD);   // gi_time
  run_gru(0, y1, BB * MM, TT, false);                                    // time GRU (batch 160)
  gemm(x_time, DD, wb_ih[1], b_ih[1], nullptr, gi, G3, NROWS, G3, DD);   // gi_fwd
  run_gru(1, y2, BB * TT, MM, false);                                    // freq fwd (batch 512)
  gemm(x_time, DD, wb_ih[2], b_ih[2], nullptr, gi, G3, NROWS, G3, DD);   // gi_bwd
  run_gru(2, y3, BB * TT, MM, true);                                     // freq bwd
  build_cat_kernel<<<(EW + 255) / 256, 256, 0, stream>>>(y1, y2, y3, gi, EW);
  gemm(gi, G3, wbWt, b_time, x_time, xt, DD, NROWS, DD, G3);             // xt = cat@Wt^T + b + res

  // ---- FrequencyDelayedStack ----
  add2_kernel<<<(EW + 255) / 256, 256, 0, stream>>>(xt, x_freq, xs, EW);
  gemm(xs, DD, wb_ih[3], b_ih[3], nullptr, gi, G3, NROWS, G3, DD);       // gi_stack
  run_gru(3, ys, BB * TT, MM, false);                                    // stack GRU
  gemm(ys, DD, wbWf, b_freq, x_freq, xf, DD, NROWS, DD, DD);             // xf
}
